// FCGF_point_att3_sft_7000_89575837925663
// MI455X (gfx1250) — matrix operand order repaired and checked by reading, not compiled, not run on hardware
//
#include <hip/hip_runtime.h>
#include <hip/hip_bf16.h>

// ---------------------------------------------------------------------------
// FCGF point-attention + FC head for MI455X (gfx1250, wave32, WMMA bf16)
// ---------------------------------------------------------------------------
#define B_SEG 256
#define P_PTS 2000
#define C_IN  32
#define N_PTS (B_SEG * P_PTS)          // 512000 points
#define INV_N (1.0f / (float)N_PTS)
#define EPS_BN 1e-5f

// GEMM tiling: C[256, Ntot] = A_bf16[256,K] * Bw_f32[Ntot,K]^T  (NT gemm)
#define GM   256      // full M per block (fw1 streamed exactly once)
#define GN   64
#define GK   32
#define GTHREADS 512  // 16 waves
#define APITCH 40     // LDS row pitch in bf16 halves (pad 32 -> 40)
#define BPITCH 40
#define SPLIT1 40     // 64000 / 40 = 1600 K per split (50 iters)
#define KCH1   1600
#define SPLIT2 8      // 1024 / 8 = 128 K per split (4 iters)
#define KCH2   128

typedef __attribute__((ext_vector_type(16))) __bf16 v16bf_t;
typedef __attribute__((ext_vector_type(8)))  float  v8f_t;

union FragU  { uint4  u[2]; v16bf_t v; };
union AccU   { v8f_t  v;    float   f[8]; };
union Pack4  { __bf16 h[4];  uint2  u; };
union Pack32 { __bf16 h[32]; uint4  u[4]; };

// ------------------------- TDM support (probe-gated) -----------------------
#ifndef USE_TDM
#if defined(__has_builtin)
#if __has_builtin(__builtin_amdgcn_tensor_load_to_lds) && \
    __has_builtin(__builtin_amdgcn_s_wait_tensorcnt)
#define USE_TDM 1
#endif
#endif
#endif
#ifndef USE_TDM
#define USE_TDM 0
#endif

#if USE_TDM
typedef unsigned int u32x4_t __attribute__((ext_vector_type(4)));
typedef int          i32x4_t __attribute__((ext_vector_type(4)));
typedef int          i32x8_t __attribute__((ext_vector_type(8)));

// DMA a 256x32 bf16 tile (row stride K elements) from global into LDS,
// padding 4 DWORDs after every 16 DWORDs (-> 40-half LDS row pitch).
__device__ inline void tdm_load_a_tile(const __bf16* gsrc, unsigned lds_addr,
                                       int K) {
  const unsigned long long ga = (unsigned long long)(size_t)gsrc;
  u32x4_t g0 = (u32x4_t)0u;
  g0[0] = 1u;                                        // count=1 (valid user D#)
  g0[1] = lds_addr;                                  // LDS byte address
  g0[2] = (unsigned)(ga & 0xffffffffu);              // global_addr[31:0]
  g0[3] = (unsigned)((ga >> 32) & 0x01ffffffu) | (2u << 30);  // [56:32]|type=2
  i32x8_t g1 = (i32x8_t)0;
  g1[0] = (1 << 16)        // data_size = 2 bytes
        | (1 << 20)        // pad_enable
        | (3 << 22)        // pad_interval: 16 DWORDs (64B of data)
        | (3 << 25);       // pad_amount:   4 DWORDs (16B of pad)
  g1[1] = (GK & 0xffff) << 16;                       // tensor_dim0 = 32
  g1[2] = (GM & 0xffff) << 16;                       // tensor_dim1 = 256
  g1[3] = (GK & 0xffff) << 16;                       // tile_dim0 = 32
  g1[4] = GM;                                        // tile_dim1 = 256
  g1[5] = K;                                         // tensor_dim0_stride
  i32x4_t z4 = (i32x4_t)0;
#if defined(__clang_major__) && (__clang_major__ >= 23)
  i32x8_t z8 = (i32x8_t)0;
  __builtin_amdgcn_tensor_load_to_lds(g0, g1, z4, z4, z8, 0);
#else
  __builtin_amdgcn_tensor_load_to_lds(g0, g1, z4, z4, 0);
#endif
}
#endif  // USE_TDM

// --------------------------- block reductions ------------------------------
// Deterministic (fixed-order) reductions; result valid in thread 0 only.
__device__ inline float block_sum(float v, float* wb) {
  #pragma unroll
  for (int o = 16; o > 0; o >>= 1) v += __shfl_down(v, o, 32);
  const int lane = threadIdx.x & 31, w = threadIdx.x >> 5;
  const int nw = blockDim.x >> 5;
  __syncthreads();                 // protect wb from previous use
  if (lane == 0) wb[w] = v;
  __syncthreads();
  float r = 0.f;
  if (threadIdx.x == 0) for (int i = 0; i < nw; ++i) r += wb[i];
  return r;
}

__device__ inline float block_max(float v, float* wb) {
  #pragma unroll
  for (int o = 16; o > 0; o >>= 1) v = fmaxf(v, __shfl_down(v, o, 32));
  const int lane = threadIdx.x & 31, w = threadIdx.x >> 5;
  const int nw = blockDim.x >> 5;
  __syncthreads();
  if (lane == 0) wb[w] = v;
  __syncthreads();
  float r = -3.4e38f;
  if (threadIdx.x == 0) for (int i = 0; i < nw; ++i) r = fmaxf(r, wb[i]);
  return r;
}

// ------------------------- attention MLP passes ----------------------------
// Pass 1: h1 = x @ w1^T  (bias cancels in BN); per-block sum/sumsq of h1.
__global__ __launch_bounds__(256) void attn_pass1(
    const float* __restrict__ x, const float* __restrict__ w1,
    float* __restrict__ p1) {
  __shared__ float sw[16 * 32];
  __shared__ float wb[8];
  const int t = threadIdx.x;
  for (int i = t; i < 512; i += 256) sw[i] = w1[i];
  __syncthreads();
  const size_t p = (size_t)blockIdx.x * 256 + t;
  float xi[32];
  const float4* xv = (const float4*)(x + p * C_IN);
  #pragma unroll
  for (int i = 0; i < 8; ++i) {
    float4 v = xv[i];
    xi[4*i] = v.x; xi[4*i+1] = v.y; xi[4*i+2] = v.z; xi[4*i+3] = v.w;
  }
  float h[16];
  #pragma unroll
  for (int c = 0; c < 16; ++c) {
    float s = 0.f;
    #pragma unroll
    for (int j = 0; j < 32; ++j) s += xi[j] * sw[c*32 + j];
    h[c] = s;
  }
  for (int c = 0; c < 16; ++c) {
    float s = block_sum(h[c], wb);
    if (t == 0) p1[(size_t)blockIdx.x * 32 + c] = s;
    float q = block_sum(h[c] * h[c], wb);
    if (t == 0) p1[(size_t)blockIdx.x * 32 + 16 + c] = q;
  }
}

// Pass 2: recompute h1, apply BN1+ReLU, h2 = h1' @ w2^T; stats of h2.
__global__ __launch_bounds__(256) void attn_pass2(
    const float* __restrict__ x, const float* __restrict__ w1,
    const float* __restrict__ g1, const float* __restrict__ be1,
    const float* __restrict__ sum1, const float* __restrict__ w2,
    float* __restrict__ p2) {
  __shared__ float sw1[512], sw2[128], sc1[16], sh1[16];
  __shared__ float wb[8];
  const int t = threadIdx.x;
  for (int i = t; i < 512; i += 256) sw1[i] = w1[i];
  if (t < 128) sw2[t] = w2[t];
  if (t < 16) {
    float m   = sum1[t] * INV_N;
    float var = sum1[16 + t] * INV_N - m * m;
    float rs  = rsqrtf(var + EPS_BN) * g1[t];
    sc1[t] = rs; sh1[t] = be1[t] - m * rs;
  }
  __syncthreads();
  const size_t p = (size_t)blockIdx.x * 256 + t;
  float xi[32];
  const float4* xv = (const float4*)(x + p * C_IN);
  #pragma unroll
  for (int i = 0; i < 8; ++i) {
    float4 v = xv[i];
    xi[4*i] = v.x; xi[4*i+1] = v.y; xi[4*i+2] = v.z; xi[4*i+3] = v.w;
  }
  float a1[16];
  #pragma unroll
  for (int c = 0; c < 16; ++c) {
    float s = 0.f;
    #pragma unroll
    for (int j = 0; j < 32; ++j) s += xi[j] * sw1[c*32 + j];
    a1[c] = fmaxf(s * sc1[c] + sh1[c], 0.f);
  }
  float h2[8];
  #pragma unroll
  for (int j = 0; j < 8; ++j) {
    float s = 0.f;
    #pragma unroll
    for (int c = 0; c < 16; ++c) s += a1[c] * sw2[j*16 + c];
    h2[j] = s;
  }
  for (int j = 0; j < 8; ++j) {
    float s = block_sum(h2[j], wb);
    if (t == 0) p2[(size_t)blockIdx.x * 16 + j] = s;
    float q = block_sum(h2[j] * h2[j], wb);
    if (t == 0) p2[(size_t)blockIdx.x * 16 + 8 + j] = q;
  }
}

// Pass 3: recompute through BN2+ReLU, score = h2' @ w3^T; store + stats.
__global__ __launch_bounds__(256) void attn_pass3(
    const float* __restrict__ x, const float* __restrict__ w1,
    const float* __restrict__ g1, const float* __restrict__ be1,
    const float* __restrict__ sum1, const float* __restrict__ w2,
    const float* __restrict__ g2, const float* __restrict__ be2,
    const float* __restrict__ sum2, const float* __restrict__ w3,
    float* __restrict__ s_pre, float* __restrict__ p3) {
  __shared__ float sw1[512], sw2[128], sw3[8];
  __shared__ float sc1[16], sh1[16], sc2[8], sh2[8];
  __shared__ float wb[8];
  const int t = threadIdx.x;
  for (int i = t; i < 512; i += 256) sw1[i] = w1[i];
  if (t < 128) sw2[t] = w2[t];
  if (t < 8)   sw3[t] = w3[t];
  if (t < 16) {
    float m   = sum1[t] * INV_N;
    float var = sum1[16 + t] * INV_N - m * m;
    float rs  = rsqrtf(var + EPS_BN) * g1[t];
    sc1[t] = rs; sh1[t] = be1[t] - m * rs;
  }
  if (t < 8) {
    float m   = sum2[t] * INV_N;
    float var = sum2[8 + t] * INV_N - m * m;
    float rs  = rsqrtf(var + EPS_BN) * g2[t];
    sc2[t] = rs; sh2[t] = be2[t] - m * rs;
  }
  __syncthreads();
  const size_t p = (size_t)blockIdx.x * 256 + t;
  float xi[32];
  const float4* xv = (const float4*)(x + p * C_IN);
  #pragma unroll
  for (int i = 0; i < 8; ++i) {
    float4 v = xv[i];
    xi[4*i] = v.x; xi[4*i+1] = v.y; xi[4*i+2] = v.z; xi[4*i+3] = v.w;
  }
  float a1[16];
  #pragma unroll
  for (int c = 0; c < 16; ++c) {
    float s = 0.f;
    #pragma unroll
    for (int j = 0; j < 32; ++j) s += xi[j] * sw1[c*32 + j];
    a1[c] = fmaxf(s * sc1[c] + sh1[c], 0.f);
  }
  float sc = 0.f;
  #pragma unroll
  for (int j = 0; j < 8; ++j) {
    float s = 0.f;
    #pragma unroll
    for (int c = 0; c < 16; ++c) s += a1[c] * sw2[j*16 + c];
    sc += fmaxf(s * sc2[j] + sh2[j], 0.f) * sw3[j];
  }
  s_pre[p] = sc;
  {
    float s = block_sum(sc, wb);
    if (t == 0) p3[(size_t)blockIdx.x * 2 + 0] = s;
    float q = block_sum(sc * sc, wb);
    if (t == 0) p3[(size_t)blockIdx.x * 2 + 1] = q;
  }
}

// Fixed-order reduction of per-block partials: out[c] = sum_g part[g][c]
__global__ void reduce_partials(const float* __restrict__ part,
                                float* __restrict__ out, int G, int C) {
  const int t = threadIdx.x;
  if (t < C) {
    float s = 0.f;
    for (int g = 0; g < G; ++g) s += part[(size_t)g * C + t];
    out[t] = s;
  }
}

// ---------------- per-segment softmax + weighted pooling (-> bf16 A) -------
__global__ __launch_bounds__(256) void softmax_pool(
    const float* __restrict__ x, const float* __restrict__ s_pre,
    const float* __restrict__ sum3, const float* __restrict__ g3,
    const float* __restrict__ be3, __bf16* __restrict__ pooled) {
  __shared__ float sc[P_PTS];
  __shared__ float wb[8];
  __shared__ float bcast;
  const int b = blockIdx.x, t = threadIdx.x;
  const float m3    = sum3[0] * INV_N;
  const float var3  = sum3[1] * INV_N - m3 * m3;
  const float scale = rsqrtf(var3 + EPS_BN) * g3[0];
  const float shift = be3[0] - m3 * scale;

  float mx = -3.4e38f;
  for (int p = t; p < P_PTS; p += 256) {
    float v = fmaxf(s_pre[(size_t)b * P_PTS + p] * scale + shift, 0.f);
    sc[p] = v;
    mx = fmaxf(mx, v);
  }
  float bm = block_max(mx, wb);
  if (t == 0) bcast = bm;
  __syncthreads();
  bm = bcast;

  float se = 0.f;
  for (int p = t; p < P_PTS; p += 256) se += expf(sc[p] - bm);
  float tot = block_sum(se, wb);
  if (t == 0) bcast = tot;
  __syncthreads();
  const float inv = 1.f / bcast;

  for (int p = t; p < P_PTS; p += 256) {
    const float a = expf(sc[p] - bm) * inv;
    const float4* xv = (const float4*)(x + ((size_t)b * P_PTS + p) * C_IN);
    Pack32 pk;
    #pragma unroll
    for (int i = 0; i < 8; ++i) {
      float4 v = xv[i];
      pk.h[4*i]   = (__bf16)(v.x * a);
      pk.h[4*i+1] = (__bf16)(v.y * a);
      pk.h[4*i+2] = (__bf16)(v.z * a);
      pk.h[4*i+3] = (__bf16)(v.w * a);
    }
    uint4* dst = (uint4*)(pooled + (size_t)b * (P_PTS * C_IN) + (size_t)p * C_IN);
    #pragma unroll
    for (int i = 0; i < 4; ++i) dst[i] = pk.u[i];
  }
}

// --------------------------- WMMA bf16 NT GEMM -----------------------------
// part[sp][m][n] = sum_{k in split sp} A[m,k] * Bw[n,k]
// A: bf16 [256,K] row-major (TDM-staged to LDS when available);
// Bw: fp32 [Ntot,K] row-major, converted to bf16 in-register while staging.
// Double-buffered LDS pipeline: one barrier per K-step.
__global__ __launch_bounds__(GTHREADS) void gemm_bf16_nt(
    const __bf16* __restrict__ A, const float* __restrict__ Bw,
    float* __restrict__ part, int Ntot, int K, int kChunk) {
  __shared__ __bf16 As[2][GM * APITCH];
  __shared__ __bf16 Bs[2][GN * BPITCH];

  const int t  = threadIdx.x;
  const int bn = blockIdx.x;           // N/64 tiles
  const int sp = blockIdx.y;           // split-K index
  const int k0 = sp * kChunk;
  int kEnd = k0 + kChunk; if (kEnd > K) kEnd = K;

  const int wave = t >> 5, lane = t & 31;
  const int wn = wave & 1, wm = wave >> 1;      // 8 x 2 wave grid
  const int m0 = wm * 32, n0 = wn * 32;
  const int l15 = lane & 15;
  const int hiA = (lane & 16) ? 8  : 0;         // A K-half offset (halves)
  const int hiB = (lane & 16) ? 8  : 0;         // B K-half offset (halves)

  // B staging: each of 512 threads owns one float4 chunk (64 rows x 8 chunks)
  const int brow = t >> 3, bc4 = (t & 7) * 4;
  const float* bsrc = Bw + (size_t)(bn * GN + brow) * K + bc4;

#if !USE_TDM
  // A staging fallback: two 8-half chunks per thread
  const int ar0 = t >> 2,          ac0 = (t & 3) * 8;
  const int ar1 = (t + 512) >> 2,  ac1 = ((t + 512) & 3) * 8;
#endif

  AccU acc[2][2];
  #pragma unroll
  for (int mi = 0; mi < 2; ++mi)
    #pragma unroll
    for (int ni = 0; ni < 2; ++ni)
      #pragma unroll
      for (int e = 0; e < 8; ++e) acc[mi][ni].f[e] = 0.f;

  // ---- prologue: stage tile k0 into buffer 0 ----
  {
#if USE_TDM
    if (t < 32) tdm_load_a_tile(A + k0, (unsigned)(size_t)&As[0][0], K);
#else
    const uint4 a0 = *(const uint4*)(A + (size_t)ar0 * K + k0 + ac0);
    const uint4 a1 = *(const uint4*)(A + (size_t)ar1 * K + k0 + ac1);
    *(uint4*)&As[0][ar0 * APITCH + ac0] = a0;
    *(uint4*)&As[0][ar1 * APITCH + ac1] = a1;
#endif
    const float4 bv = *(const float4*)(bsrc + k0);
    Pack4 pk;
    pk.h[0] = (__bf16)bv.x; pk.h[1] = (__bf16)bv.y;
    pk.h[2] = (__bf16)bv.z; pk.h[3] = (__bf16)bv.w;
    *(uint2*)&Bs[0][brow * BPITCH + bc4] = pk.u;
#if USE_TDM
    if (t < 32) __builtin_amdgcn_s_wait_tensorcnt(0);
#endif
    __syncthreads();
  }

  int cur = 0;
  for (int kb = k0; kb < kEnd; kb += GK) {
    const int kn = kb + GK;
    const bool more = kn < kEnd;

    // ---- start staging tile kn into buffer cur^1 ----
    float4 bv;
#if !USE_TDM
    uint4 an0, an1;
#endif
    if (more) {
#if USE_TDM
      if (t < 32) tdm_load_a_tile(A + kn, (unsigned)(size_t)&As[cur ^ 1][0], K);
#else
      an0 = *(const uint4*)(A + (size_t)ar0 * K + kn + ac0);
      an1 = *(const uint4*)(A + (size_t)ar1 * K + kn + ac1);
#endif
      bv = *(const float4*)(bsrc + kn);
      if (kn + GK < kEnd) __builtin_prefetch(bsrc + kn + GK, 0, 1);
    }

    // ---- compute on buffer cur ----
    FragU a[2], bfr[2];
    #pragma unroll
    for (int mi = 0; mi < 2; ++mi) {
      const int r = (m0 + mi * 16 + l15) * APITCH;
      a[mi].u[0] = *(const uint4*)&As[cur][r + hiA];        // K hiA..hiA+7
      a[mi].u[1] = *(const uint4*)&As[cur][r + hiA + 16];   // K hiA+16..+23
    }
    #pragma unroll
    for (int ni = 0; ni < 2; ++ni) {
      const int r = (n0 + ni * 16 + l15) * BPITCH;
      bfr[ni].u[0] = *(const uint4*)&Bs[cur][r + hiB];      // K hiB..hiB+7
      bfr[ni].u[1] = *(const uint4*)&Bs[cur][r + hiB + 16]; // K hiB+16..+23
    }
    #pragma unroll
    for (int mi = 0; mi < 2; ++mi)
      #pragma unroll
      for (int ni = 0; ni < 2; ++ni)
        acc[mi][ni].v = __builtin_amdgcn_wmma_f32_16x16x32_bf16(
            false, a[mi].v, false, bfr[ni].v,
            (short)0, acc[mi][ni].v, false, false);

    // ---- finish staging tile kn ----
    if (more) {
#if !USE_TDM
      *(uint4*)&As[cur ^ 1][ar0 * APITCH + ac0] = an0;
      *(uint4*)&As[cur ^ 1][ar1 * APITCH + ac1] = an1;
#endif
      Pack4 pk;
      pk.h[0] = (__bf16)bv.x; pk.h[1] = (__bf16)bv.y;
      pk.h[2] = (__bf16)bv.z; pk.h[3] = (__bf16)bv.w;
      *(uint2*)&Bs[cur ^ 1][brow * BPITCH + bc4] = pk.u;
#if USE_TDM
      if (t < 32) __builtin_amdgcn_s_wait_tensorcnt(0);
#endif
    }
    __syncthreads();
    cur ^= 1;
  }

  // ---- epilogue: write split-K partials ----
  float* dst = part + (size_t)sp * 256 * Ntot;
  const int mhi = (lane & 16) ? 8 : 0;
  #pragma unroll
  for (int mi = 0; mi < 2; ++mi)
    #pragma unroll
    for (int ni = 0; ni < 2; ++ni)
      #pragma unroll
      for (int e = 0; e < 8; ++e) {
        const int m = m0 + mi * 16 + e + mhi;
        const int n = bn * GN + n0 + ni * 16 + l15;
        dst[(size_t)m * Ntot + n] = acc[mi][ni].f[e];
      }
}

// split-K reduce (fixed order) + column BN stats over the 256 rows
__global__ __launch_bounds__(256) void colstats_reduce(
    const float* __restrict__ part, int S, int Ntot,
    float* __restrict__ r, float* __restrict__ mean, float* __restrict__ rstd) {
  const int c = blockIdx.x, t = threadIdx.x;   // t = row (M=256)
  __shared__ float wb[8];
  float v = 0.f;
  for (int s = 0; s < S; ++s) v += part[((size_t)s * 256 + t) * Ntot + c];
  r[(size_t)t * Ntot + c] = v;
  float s1 = block_sum(v, wb);
  float s2 = block_sum(v * v, wb);
  if (t == 0) {
    float m   = s1 * (1.0f / 256.0f);
    float var = s2 * (1.0f / 256.0f) - m * m;
    mean[c] = m;
    rstd[c] = rsqrtf(var + EPS_BN);
  }
}

// BN + ReLU -> bf16 activations (input of next WMMA gemm)
__global__ __launch_bounds__(256) void apply_bn_bf16(
    const float* __restrict__ r, const float* __restrict__ mean,
    const float* __restrict__ rstd, const float* __restrict__ g,
    const float* __restrict__ be, __bf16* __restrict__ out, int Ntot) {
  const int i = blockIdx.x * 256 + threadIdx.x;
  const int c = i % Ntot;
  const float y = (r[i] - mean[c]) * rstd[c] * g[c] + be[c];
  out[i] = (__bf16)fmaxf(y, 0.f);
}

// BN + ReLU + row L2-normalize -> output [256,256]
__global__ __launch_bounds__(256) void final_norm(
    const float* __restrict__ r2, const float* __restrict__ mean,
    const float* __restrict__ rstd, const float* __restrict__ g,
    const float* __restrict__ be, float* __restrict__ out) {
  const int m = blockIdx.x, c = threadIdx.x;
  __shared__ float wb[8];
  __shared__ float nrm;
  float y = (r2[m * 256 + c] - mean[c]) * rstd[c] * g[c] + be[c];
  y = fmaxf(y, 0.f);
  float ss = block_sum(y * y, wb);
  if (c == 0) nrm = fmaxf(sqrtf(ss), 1e-12f);
  __syncthreads();
  out[m * 256 + c] = y / nrm;
}

// ---------------------------------------------------------------------------
extern "C" void kernel_launch(void* const* d_in, const int* in_sizes, int n_in,
                              void* d_out, int out_size, void* d_ws, size_t ws_size,
                              hipStream_t stream) {
  const float* x    = (const float*)d_in[0];
  // d_in[1] = length (always P, unused). All linear biases cancel in BN.
  const float* w1   = (const float*)d_in[2];
  const float* g1   = (const float*)d_in[4];
  const float* be1  = (const float*)d_in[5];
  const float* w2   = (const float*)d_in[6];
  const float* g2   = (const float*)d_in[8];
  const float* be2  = (const float*)d_in[9];
  const float* w3   = (const float*)d_in[10];
  const float* g3   = (const float*)d_in[12];
  const float* be3  = (const float*)d_in[13];
  const float* fw1  = (const float*)d_in[14];
  const float* fg1  = (const float*)d_in[16];
  const float* fbe1 = (const float*)d_in[17];
  const float* fw2  = (const float*)d_in[18];
  const float* fg2  = (const float*)d_in[20];
  const float* fbe2 = (const float*)d_in[21];
  float* out = (float*)d_out;

  // workspace carve (256B aligned regions; every byte we read is written first)
  size_t off = 0;
  auto carve = [&](size_t bytes) -> char* {
    char* p = (char*)d_ws + off;
    off += (bytes + 255) & ~(size_t)255;
    return p;
  };
  float*  p1     = (float*)carve((size_t)2000 * 32 * 4);
  float*  p2     = (float*)carve((size_t)2000 * 16 * 4);
  float*  p3     = (float*)carve((size_t)2000 * 2 * 4);
  float*  sum1   = (float*)carve(32 * 4);
  float*  sum2   = (float*)carve(16 * 4);
  float*  sum3   = (float*)carve(2 * 4);
  float*  s_pre  = (float*)carve((size_t)N_PTS * 4);
  __bf16* pooled = (__bf16*)carve((size_t)B_SEG * 64000 * 2);
  float*  part1  = (float*)carve((size_t)SPLIT1 * 256 * 1024 * 4);
  float*  r1     = (float*)carve((size_t)256 * 1024 * 4);
  float*  mean1  = (float*)carve(1024 * 4);
  float*  rstd1  = (float*)carve(1024 * 4);
  __bf16* a1bf   = (__bf16*)carve((size_t)256 * 1024 * 2);
  float*  part2  = (float*)carve((size_t)SPLIT2 * 256 * 256 * 4);
  float*  r2     = (float*)carve((size_t)256 * 256 * 4);
  float*  mean2  = (float*)carve(256 * 4);
  float*  rstd2  = (float*)carve(256 * 4);

  // attention MLP: 3 passes + deterministic stat reductions
  attn_pass1<<<2000, 256, 0, stream>>>(x, w1, p1);
  reduce_partials<<<1, 64, 0, stream>>>(p1, sum1, 2000, 32);
  attn_pass2<<<2000, 256, 0, stream>>>(x, w1, g1, be1, sum1, w2, p2);
  reduce_partials<<<1, 64, 0, stream>>>(p2, sum2, 2000, 16);
  attn_pass3<<<2000, 256, 0, stream>>>(x, w1, g1, be1, sum1, w2, g2, be2, sum2,
                                       w3, s_pre, p3);
  reduce_partials<<<1, 64, 0, stream>>>(p3, sum3, 2000, 2);

  // per-segment softmax + pooling -> bf16 A matrix
  softmax_pool<<<B_SEG, 256, 0, stream>>>(x, s_pre, sum3, g3, be3, pooled);

  // FC1: [256,64000] x [64000,1024]^T via WMMA bf16, split-K=40
  gemm_bf16_nt<<<dim3(1024 / GN, SPLIT1), GTHREADS, 0, stream>>>(
      pooled, fw1, part1, 1024, 64000, KCH1);
  colstats_reduce<<<1024, 256, 0, stream>>>(part1, SPLIT1, 1024, r1, mean1, rstd1);
  apply_bn_bf16<<<(256 * 1024) / 256, 256, 0, stream>>>(r1, mean1, rstd1, fg1,
                                                        fbe1, a1bf, 1024);

  // FC2: [256,1024] x [1024,256]^T via WMMA bf16, split-K=8
  gemm_bf16_nt<<<dim3(256 / GN, SPLIT2), GTHREADS, 0, stream>>>(
      a1bf, fw2, part2, 256, 1024, KCH2);
  colstats_reduce<<<256, 256, 0, stream>>>(part2, SPLIT2, 256, r2, mean2, rstd2);

  // BN + ReLU + L2 normalize rows -> output
  final_norm<<<B_SEG, 256, 0, stream>>>(r2, mean2, rstd2, fg2, fbe2, out);
}
